// LatentAdaCUT_4294967296489
// MI455X (gfx1250) — matrix operand order repaired and checked by reading, not compiled, not run on hardware
//
#include <hip/hip_runtime.h>
#include <hip/hip_bf16.h>

// ---------- CDNA5 WMMA types ----------
typedef __attribute__((ext_vector_type(16))) __bf16 v16bf;
typedef __attribute__((ext_vector_type(8)))  __bf16 v8bf;
typedef __attribute__((ext_vector_type(8)))  float  v8f;

__device__ __forceinline__ unsigned short f2bf(float f) {
    __bf16 h = (__bf16)f;
    return __builtin_bit_cast(unsigned short, h);
}
__device__ __forceinline__ float bf2f(unsigned short u) {
    return (float)__builtin_bit_cast(__bf16, u);
}

// ---------- problem constants ----------
#define BATCH 16
#define CCH   256
#define HW    4096
#define MROWS (BATCH * HW)       // 65536
#define NTOK  64
#define NHEAD 4
#define HD    64
#define EPSV  1e-5f

// ---------- workspace layout (all offsets multiple of 256 bytes) ----------
#define OFF_GP   ((size_t)0)                                  // 16*512 f32
#define OFF_S    (OFF_GP  + (size_t)16*512*4)                 // 16*256 f32
#define OFF_GNM  (OFF_S   + (size_t)16*256*4)                 // 64 f32
#define OFF_GNR  (OFF_GNM + (size_t)256)                      // 64 f32
#define OFF_POS  (OFF_GNR + (size_t)256)                      // 4096*256 f32
#define OFF_WQT  (OFF_POS + (size_t)HW*CCH*4)                 // 256*256 bf16
#define OFF_WOT  (OFF_WQT + (size_t)256*256*2)
#define OFF_WF1T (OFF_WOT + (size_t)256*256*2)                // 512*256 bf16
#define OFF_WF2T (OFF_WF1T + (size_t)512*256*2)               // 256*512 bf16
#define OFF_K    (OFF_WF2T + (size_t)256*512*2)               // 16*64*256 bf16
#define OFF_V    (OFF_K    + (size_t)16*64*256*2)
#define OFF_QIN  (OFF_V    + (size_t)16*64*256*2)             // 65536*256 bf16 (reused as LN(o))
#define OFF_Q    (OFF_QIN  + (size_t)MROWS*256*2)             // 65536*256 bf16
#define OFF_SC   (OFF_Q    + (size_t)MROWS*256*2)             // 65536*256 bf16
#define OFF_O    (OFF_SC   + (size_t)MROWS*256*2)             // 65536*256 bf16
#define OFF_H    (OFF_O    + (size_t)MROWS*256*2)             // 65536*512 bf16
#define OFF_F    (OFF_H    + (size_t)MROWS*512*2)             // 65536*256 f32

// ================= small prep kernels =================

__global__ __launch_bounds__(256) void style_kernel(
    const float* __restrict__ style, const float* __restrict__ Wg,
    const float* __restrict__ bgv, const float* __restrict__ Ws,
    const float* __restrict__ bsv, float* __restrict__ gp, float* __restrict__ sout)
{
    int b = blockIdx.x, t = threadIdx.x;
    const float* st = style + b * 512;
    for (int j = 0; j < 2; ++j) {
        int col = j * 256 + t;
        float acc = bgv[col];
        for (int k2 = 0; k2 < 512; ++k2) acc += st[k2] * Wg[(size_t)k2 * 512 + col];
        gp[b * 512 + col] = acc;
    }
    float acc = bsv[t];
    for (int k2 = 0; k2 < 512; ++k2) acc += st[k2] * Ws[(size_t)k2 * 256 + t];
    sout[b * 256 + t] = acc;
}

__global__ __launch_bounds__(256) void gn_stats_kernel(
    const float* __restrict__ x, float* __restrict__ mean, float* __restrict__ rstd)
{
    int bg = blockIdx.x, t = threadIdx.x;
    const float4* p = (const float4*)(x + (size_t)bg * 64 * HW);
    float s = 0.f, s2 = 0.f;
    for (int i = t; i < (64 * HW) / 4; i += 256) {
        float4 v = p[i];
        s  += v.x + v.y + v.z + v.w;
        s2 += v.x * v.x + v.y * v.y + v.z * v.z + v.w * v.w;
    }
    __shared__ float r1[256], r2[256];
    r1[t] = s; r2[t] = s2; __syncthreads();
    for (int st = 128; st; st >>= 1) {
        if (t < st) { r1[t] += r1[t + st]; r2[t] += r2[t + st]; }
        __syncthreads();
    }
    if (t == 0) {
        float m = r1[0] / (64.f * HW);
        float var = r2[0] / (64.f * HW) - m * m;
        mean[bg] = m; rstd[bg] = rsqrtf(var + EPSV);
    }
}

__global__ __launch_bounds__(256) void pos_kernel(
    const float* __restrict__ Wp1, const float* __restrict__ bp1,
    const float* __restrict__ Wp2, const float* __restrict__ bp2,
    float* __restrict__ pos)
{
    int p = blockIdx.x, t = threadIdx.x;
    float gy = -1.f + 2.f * (float)(p >> 6) / 63.f;
    float gx = -1.f + 2.f * (float)(p & 63) / 63.f;
    __shared__ float hl[256];
    float h = gx * Wp1[t] + gy * Wp1[256 + t] + bp1[t];
    h = h / (1.f + __expf(-h));      // silu
    hl[t] = h; __syncthreads();
    float acc = bp2[t];
    for (int c = 0; c < 256; ++c) acc += hl[c] * Wp2[(size_t)c * 256 + t];
    pos[(size_t)p * 256 + t] = acc;
}

__global__ __launch_bounds__(256) void tok_kv_kernel(
    const float* __restrict__ tokens, const float* __restrict__ sbuf,
    const float* __restrict__ g, const float* __restrict__ bln,
    const float* __restrict__ Wk, const float* __restrict__ bk,
    const float* __restrict__ Wv, const float* __restrict__ bv,
    unsigned short* __restrict__ kout, unsigned short* __restrict__ vout)
{
    int b = blockIdx.x >> 6, tok = blockIdx.x & 63, t = threadIdx.x;
    __shared__ float tl[256], r1[256], r2[256];
    float v = tokens[tok * 256 + t] + sbuf[b * 256 + t];
    r1[t] = v; r2[t] = v * v; __syncthreads();
    for (int st = 128; st; st >>= 1) {
        if (t < st) { r1[t] += r1[t + st]; r2[t] += r2[t + st]; }
        __syncthreads();
    }
    float m = r1[0] / 256.f;
    float rs = rsqrtf(r2[0] / 256.f - m * m + EPSV);
    tl[t] = (v - m) * rs * g[t] + bln[t];
    __syncthreads();
    float ak = bk[t], av = bv[t];
    for (int c = 0; c < 256; ++c) {
        float tv = tl[c];
        ak += tv * Wk[(size_t)c * 256 + t];
        av += tv * Wv[(size_t)c * 256 + t];
    }
    size_t o = ((size_t)b * 64 + tok) * 256 + t;
    kout[o] = f2bf(ak); vout[o] = f2bf(av);
}

__global__ void wtrans_kernel(const float* __restrict__ W, unsigned short* __restrict__ Wt,
                              int K, int N)
{
    int n = blockIdx.x;
    for (int k2 = threadIdx.x; k2 < K; k2 += blockDim.x)
        Wt[(size_t)n * K + k2] = f2bf(W[(size_t)k2 * N + n]);
}

__global__ __launch_bounds__(256) void qin_kernel(
    const float* __restrict__ x, const float* __restrict__ pos,
    const float* __restrict__ gnm, const float* __restrict__ gnr,
    const float* __restrict__ g, const float* __restrict__ bln,
    unsigned short* __restrict__ qin)
{
    __shared__ float tile[32][260];
    int b = blockIdx.x >> 7, p0 = (blockIdx.x & 127) * 32, t = threadIdx.x;
    int pl = t & 31, c_off = t >> 5;
    for (int cc = c_off; cc < 256; cc += 8) {
        int gi = b * 4 + (cc >> 6);
        float xv = x[((size_t)b * 256 + cc) * HW + p0 + pl];
        tile[pl][cc] = (xv - gnm[gi]) * gnr[gi];
    }
    __syncthreads();
    int p = t >> 3, cb = (t & 7) * 32;
    const float* pr = pos + (size_t)(p0 + p) * 256;
    float s = 0.f, s2 = 0.f;
    for (int i = 0; i < 32; ++i) {
        float v = tile[p][cb + i] + pr[cb + i];
        tile[p][cb + i] = v; s += v; s2 += v * v;
    }
    s += __shfl_xor(s, 1); s2 += __shfl_xor(s2, 1);
    s += __shfl_xor(s, 2); s2 += __shfl_xor(s2, 2);
    s += __shfl_xor(s, 4); s2 += __shfl_xor(s2, 4);
    float m = s * (1.f / 256.f);
    float rs = rsqrtf(s2 * (1.f / 256.f) - m * m + EPSV);
    unsigned short* dst = qin + ((size_t)b * HW + p0 + p) * 256;
    for (int i = 0; i < 32; ++i) {
        int c = cb + i;
        dst[c] = f2bf((tile[p][c] - m) * rs * g[c] + bln[c]);
    }
}

__global__ __launch_bounds__(256) void ln_rows_kernel(
    const unsigned short* __restrict__ in, const float* __restrict__ g,
    const float* __restrict__ bln, unsigned short* __restrict__ out)
{
    int w = threadIdx.x >> 5, lane = threadIdx.x & 31;
    size_t row = (size_t)blockIdx.x * 8 + w;
    const unsigned short* p = in + row * 256 + lane * 8;
    float vals[8], s = 0.f, s2 = 0.f;
    for (int i = 0; i < 8; ++i) { float v = bf2f(p[i]); vals[i] = v; s += v; s2 += v * v; }
    for (int off = 16; off; off >>= 1) { s += __shfl_xor(s, off); s2 += __shfl_xor(s2, off); }
    float m = s * (1.f / 256.f);
    float rs = rsqrtf(s2 * (1.f / 256.f) - m * m + EPSV);
    unsigned short* q = out + row * 256 + lane * 8;
    for (int i = 0; i < 8; ++i) {
        int c = lane * 8 + i;
        q[i] = f2bf((vals[i] - m) * rs * g[c] + bln[c]);
    }
}

// ================= bf16 WMMA GEMM (2x2 wave tile, async double-buffered LDS) ======
// out[M x N] = A[M x K](bf16) @ Wt[N x K](bf16)^T + bias; flags:
//   bit0: silu, bit1: bf16 output, bit2: add bf16 residual (same [row*N+col])
// block tile: 64M x 128N, 8 waves (2x4), each wave 32x32 via 4 accumulators.
// Weight chunks (128N x 64K bf16) staged via GLOBAL_LOAD_ASYNC_TO_LDS_B128,
// double-buffered, tracked with ASYNCcnt.
__global__ __launch_bounds__(256) void gemm_bf16_kernel(
    const unsigned short* __restrict__ A, const unsigned short* __restrict__ Wt,
    const float* __restrict__ bias, const unsigned short* __restrict__ resid,
    float* __restrict__ outF, unsigned short* __restrict__ outB,
    int Ndim, int Kdim, int flags)
{
    __shared__ unsigned short Bt[2][128][72];
    int m0 = blockIdx.x * 64, n0 = blockIdx.y * 128;
    int tid = threadIdx.x, w = tid >> 5, lane = tid & 31;
    int wm = w >> 2, wn = w & 3;                 // 2 x 4 wave grid
    int kbase = (lane >> 4) * 8, khalf = (lane >> 4) * 16, lr = lane & 15;
    const unsigned short* Arow0 = A + (size_t)(m0 + 32 * wm + lr) * Kdim;
    const unsigned short* Arow1 = Arow0 + (size_t)16 * Kdim;
    int bn0 = 32 * wn + lr, bn1 = bn0 + 16;      // LDS weight rows for this wave
    // async staging: thread moves 64B of one weight row per chunk
    int ldr = tid >> 1, ldc = (tid & 1) * 32;
    const unsigned short* gsrc = Wt + (size_t)(n0 + ldr) * Kdim + ldc;

    v8f acc[2][2];
    v8f vzero = {};
#pragma unroll
    for (int mi = 0; mi < 2; ++mi)
#pragma unroll
        for (int ni = 0; ni < 2; ++ni) acc[mi][ni] = vzero;

    // ---- prologue: async-stage chunk 0 into buffer 0 ----
    {
        unsigned dst = (unsigned)(size_t)&Bt[0][ldr][ldc];          // LDS byte offset
        unsigned long long src = (unsigned long long)gsrc;
        asm volatile(
            "global_load_async_to_lds_b128 %0, %1, off\n\t"
            "global_load_async_to_lds_b128 %0, %1, off offset:16\n\t"
            "global_load_async_to_lds_b128 %0, %1, off offset:32\n\t"
            "global_load_async_to_lds_b128 %0, %1, off offset:48"
            :: "v"(dst), "v"(src) : "memory");
        asm volatile("s_wait_asynccnt 0" ::: "memory");
        __syncthreads();
    }

    int nch = Kdim >> 6, cur = 0;
    for (int ch = 0; ch < nch; ++ch) {
        int k0 = ch * 64;
        if (ch + 1 < nch) {                      // prefetch next chunk into other buffer
            unsigned dst = (unsigned)(size_t)&Bt[cur ^ 1][ldr][ldc];
            unsigned long long src = (unsigned long long)(gsrc + k0 + 64);
            asm volatile(
                "global_load_async_to_lds_b128 %0, %1, off\n\t"
                "global_load_async_to_lds_b128 %0, %1, off offset:16\n\t"
                "global_load_async_to_lds_b128 %0, %1, off offset:32\n\t"
                "global_load_async_to_lds_b128 %0, %1, off offset:48"
                :: "v"(dst), "v"(src) : "memory");
        }
        __builtin_prefetch(Arow0 + k0 + 64, 0, 1);   // global_prefetch_b8 (A stream)
        __builtin_prefetch(Arow1 + k0 + 64, 0, 1);
#pragma unroll
        for (int kk = 0; kk < 64; kk += 32) {
            v16bf a[2], bm[2];
            {
                const v8bf a0lo = *(const v8bf*)(Arow0 + k0 + kk + kbase);
                const v8bf a0hi = *(const v8bf*)(Arow0 + k0 + kk + 16 + kbase);
                const v8bf a1lo = *(const v8bf*)(Arow1 + k0 + kk + kbase);
                const v8bf a1hi = *(const v8bf*)(Arow1 + k0 + kk + 16 + kbase);
                const v8bf b0lo = *(const v8bf*)(&Bt[cur][bn0][kk + kbase]);
                const v8bf b0hi = *(const v8bf*)(&Bt[cur][bn0][kk + 16 + kbase]);
                const v8bf b1lo = *(const v8bf*)(&Bt[cur][bn1][kk + kbase]);
                const v8bf b1hi = *(const v8bf*)(&Bt[cur][bn1][kk + 16 + kbase]);
#pragma unroll
                for (int i = 0; i < 8; ++i) {
                    a[0][i] = a0lo[i]; a[0][8 + i] = a0hi[i];
                    a[1][i] = a1lo[i]; a[1][8 + i] = a1hi[i];
                    bm[0][i] = b0lo[i]; bm[0][8 + i] = b0hi[i];
                    bm[1][i] = b1lo[i]; bm[1][8 + i] = b1hi[i];
                }
            }
#pragma unroll
            for (int mi = 0; mi < 2; ++mi)
#pragma unroll
                for (int ni = 0; ni < 2; ++ni)
                    acc[mi][ni] = __builtin_amdgcn_wmma_f32_16x16x32_bf16(
                        false, a[mi], false, bm[ni], (short)0, acc[mi][ni], false, false);
        }
        asm volatile("s_wait_asynccnt 0" ::: "memory");   // next chunk landed
        __syncthreads();                                  // all waves done read+stage
        cur ^= 1;
    }

    // ---- epilogue ----
    int col0 = n0 + 32 * wn + lr;
#pragma unroll
    for (int ni = 0; ni < 2; ++ni) {
        int col = col0 + ni * 16;
        float bc = bias[col];
#pragma unroll
        for (int mi = 0; mi < 2; ++mi) {
#pragma unroll
            for (int j = 0; j < 8; ++j) {
                int row = m0 + 32 * wm + mi * 16 + 8 * (lane >> 4) + j;
                size_t idx = (size_t)row * Ndim + col;
                float v = acc[mi][ni][j] + bc;
                if (flags & 1) v = v / (1.f + __expf(-v));     // silu
                if (flags & 4) v += bf2f(resid[idx]);          // residual add
                if (flags & 2) outB[idx] = f2bf(v); else outF[idx] = v;
            }
        }
    }
}

// ================= attention (WMMA QK^T, softmax, WMMA attn*V) =================
__global__ __launch_bounds__(128) void attn_kernel(
    const unsigned short* __restrict__ q, const unsigned short* __restrict__ kbuf,
    const unsigned short* __restrict__ vbuf, unsigned short* __restrict__ sc)
{
    __shared__ unsigned short kl[64][72];       // k[tok][d]
    __shared__ unsigned short vT[64][72];       // v^T[d][tok]
    __shared__ float scor[4][16][64];
    __shared__ unsigned short at[4][16][72];
    int b = blockIdx.y >> 2, h = blockIdx.y & 3, t = threadIdx.x;
    {
        int tok = t >> 1, half = t & 1;
        const unsigned short* ks = kbuf + ((size_t)(b * 64 + tok) * 256 + h * 64 + half * 32);
        const unsigned short* vs = vbuf + ((size_t)(b * 64 + tok) * 256 + h * 64 + half * 32);
        const uint4* k4 = (const uint4*)ks;
        uint4* d4 = (uint4*)&kl[tok][half * 32];
        d4[0] = k4[0]; d4[1] = k4[1]; d4[2] = k4[2]; d4[3] = k4[3];
        for (int i = 0; i < 32; ++i) vT[half * 32 + i][tok] = vs[i];
    }
    __syncthreads();
    int w = t >> 5, lane = t & 31;
    int q0 = blockIdx.x * 64 + w * 16;
    int kbase = (lane >> 4) * 8, khalf = (lane >> 4) * 16, lr = lane & 15;
    const unsigned short* qrow = q + ((size_t)(b * HW + q0 + lr) * 256 + h * 64);
    v8f zero = {};
    v8f accs[4] = { zero, zero, zero, zero };
#pragma unroll
    for (int kk = 0; kk < 64; kk += 32) {
        v16bf a;
        const v8bf alo = *(const v8bf*)(qrow + kk + kbase);
        const v8bf ahi = *(const v8bf*)(qrow + kk + 16 + kbase);
#pragma unroll
        for (int i = 0; i < 8; ++i) { a[i] = alo[i]; a[8 + i] = ahi[i]; }
#pragma unroll
        for (int nt = 0; nt < 4; ++nt) {
            v16bf bm;
            const v8bf blo = *(const v8bf*)(&kl[nt * 16 + lr][kk + kbase]);
            const v8bf bhi = *(const v8bf*)(&kl[nt * 16 + lr][kk + 16 + kbase]);
#pragma unroll
            for (int i = 0; i < 8; ++i) { bm[i] = blo[i]; bm[8 + i] = bhi[i]; }
            accs[nt] = __builtin_amdgcn_wmma_f32_16x16x32_bf16(
                false, a, false, bm, (short)0, accs[nt], false, false);
        }
    }
    const float ascale = 0.25f;   // SHARPEN / sqrt(HEAD_DIM) = 2/8
#pragma unroll
    for (int nt = 0; nt < 4; ++nt)
#pragma unroll
        for (int j = 0; j < 8; ++j)
            scor[w][8 * (lane >> 4) + j][nt * 16 + lr] = accs[nt][j] * ascale;
    if (lane < 16) {                      // per-row softmax over 64 tokens
        float mx = -1e30f;
        for (int tk = 0; tk < 64; ++tk) mx = fmaxf(mx, scor[w][lane][tk]);
        float sum = 0.f;
        for (int tk = 0; tk < 64; ++tk) {
            float e = __expf(scor[w][lane][tk] - mx);
            scor[w][lane][tk] = e; sum += e;
        }
        float inv = 1.f / sum;
        for (int tk = 0; tk < 64; ++tk) at[w][lane][tk] = f2bf(scor[w][lane][tk] * inv);
    }
    v8f o4[4] = { zero, zero, zero, zero };
#pragma unroll
    for (int kk = 0; kk < 64; kk += 32) {
        v16bf a;
        const v8bf alo = *(const v8bf*)(&at[w][lr][kk + kbase]);
        const v8bf ahi = *(const v8bf*)(&at[w][lr][kk + 16 + kbase]);
#pragma unroll
        for (int i = 0; i < 8; ++i) { a[i] = alo[i]; a[8 + i] = ahi[i]; }
#pragma unroll
        for (int nt = 0; nt < 4; ++nt) {
            v16bf bm;
            const v8bf blo = *(const v8bf*)(&vT[nt * 16 + lr][kk + kbase]);
            const v8bf bhi = *(const v8bf*)(&vT[nt * 16 + lr][kk + 16 + kbase]);
#pragma unroll
            for (int i = 0; i < 8; ++i) { bm[i] = blo[i]; bm[8 + i] = bhi[i]; }
            o4[nt] = __builtin_amdgcn_wmma_f32_16x16x32_bf16(
                false, a, false, bm, (short)0, o4[nt], false, false);
        }
    }
    unsigned short* out = sc + ((size_t)(b * HW + q0) * 256 + h * 64);
#pragma unroll
    for (int nt = 0; nt < 4; ++nt)
#pragma unroll
        for (int j = 0; j < 8; ++j) {
            int m = 8 * (lane >> 4) + j;
            out[(size_t)m * 256 + nt * 16 + lr] = f2bf(o4[nt][j]);
        }
}

// final: out[b][c][p] = normed*scale + shift + F[b*HW+p][c]*gamma
__global__ __launch_bounds__(256) void out_kernel(
    const float* __restrict__ x, const float* __restrict__ F,
    const float* __restrict__ gnm, const float* __restrict__ gnr,
    const float* __restrict__ gp, const float* __restrict__ gamma,
    float* __restrict__ out)
{
    __shared__ float tile[32][260];
    int b = blockIdx.x >> 7, p0 = (blockIdx.x & 127) * 32, t = threadIdx.x;
    int p = t >> 3, cb = (t & 7) * 32;
    const float* Fr = F + ((size_t)b * HW + p0 + p) * 256;
    for (int i = 0; i < 32; ++i) tile[p][cb + i] = Fr[cb + i];
    __syncthreads();
    int pl = t & 31, c_off = t >> 5;
    for (int cc = c_off; cc < 256; cc += 8) {
        int gi = b * 4 + (cc >> 6);
        float xv = x[((size_t)b * 256 + cc) * HW + p0 + pl];
        float nv = (xv - gnm[gi]) * gnr[gi];
        float r = nv * gp[b * 512 + cc] + gp[b * 512 + 256 + cc] + tile[pl][cc] * gamma[cc];
        out[((size_t)b * 256 + cc) * HW + p0 + pl] = r;
    }
}

// ================= host launcher =================
extern "C" void kernel_launch(void* const* d_in, const int* in_sizes, int n_in,
                              void* d_out, int out_size, void* d_ws, size_t ws_size,
                              hipStream_t stream)
{
    (void)in_sizes; (void)n_in; (void)out_size; (void)ws_size;
    const float* x     = (const float*)d_in[0];
    const float* style = (const float*)d_in[1];
    const float* Wg    = (const float*)d_in[2];
    const float* bg    = (const float*)d_in[3];
    const float* toks  = (const float*)d_in[4];
    const float* Ws    = (const float*)d_in[5];
    const float* bs    = (const float*)d_in[6];
    const float* Wp1   = (const float*)d_in[7];
    const float* bp1   = (const float*)d_in[8];
    const float* Wp2   = (const float*)d_in[9];
    const float* bp2   = (const float*)d_in[10];
    const float* ln_t_g = (const float*)d_in[11];
    const float* ln_t_b = (const float*)d_in[12];
    const float* ln_q_g = (const float*)d_in[13];
    const float* ln_q_b = (const float*)d_in[14];
    const float* ln_f_g = (const float*)d_in[15];
    const float* ln_f_b = (const float*)d_in[16];
    const float* Wq = (const float*)d_in[17]; const float* bq = (const float*)d_in[18];
    const float* Wk = (const float*)d_in[19]; const float* bk = (const float*)d_in[20];
    const float* Wv = (const float*)d_in[21]; const float* bv = (const float*)d_in[22];
    const float* Wo = (const float*)d_in[23]; const float* bo = (const float*)d_in[24];
    const float* Wf1 = (const float*)d_in[25]; const float* bf1 = (const float*)d_in[26];
    const float* Wf2 = (const float*)d_in[27]; const float* bf2 = (const float*)d_in[28];
    const float* gamma = (const float*)d_in[29];

    char* ws = (char*)d_ws;
    float* gp   = (float*)(ws + OFF_GP);
    float* sbuf = (float*)(ws + OFF_S);
    float* gnm  = (float*)(ws + OFF_GNM);
    float* gnr  = (float*)(ws + OFF_GNR);
    float* pos  = (float*)(ws + OFF_POS);
    unsigned short* WQT  = (unsigned short*)(ws + OFF_WQT);
    unsigned short* WOT  = (unsigned short*)(ws + OFF_WOT);
    unsigned short* WF1T = (unsigned short*)(ws + OFF_WF1T);
    unsigned short* WF2T = (unsigned short*)(ws + OFF_WF2T);
    unsigned short* kb   = (unsigned short*)(ws + OFF_K);
    unsigned short* vb   = (unsigned short*)(ws + OFF_V);
    unsigned short* qin  = (unsigned short*)(ws + OFF_QIN);
    unsigned short* qbuf = (unsigned short*)(ws + OFF_Q);
    unsigned short* scb  = (unsigned short*)(ws + OFF_SC);
    unsigned short* obuf = (unsigned short*)(ws + OFF_O);
    unsigned short* hln  = (unsigned short*)(ws + OFF_QIN);   // reuse q_in buffer
    unsigned short* hmid = (unsigned short*)(ws + OFF_H);
    float* Fbuf = (float*)(ws + OFF_F);

    // weight prep
    wtrans_kernel<<<256, 256, 0, stream>>>(Wq,  WQT,  256, 256);
    wtrans_kernel<<<256, 256, 0, stream>>>(Wo,  WOT,  256, 256);
    wtrans_kernel<<<512, 256, 0, stream>>>(Wf1, WF1T, 256, 512);
    wtrans_kernel<<<256, 256, 0, stream>>>(Wf2, WF2T, 512, 256);

    style_kernel<<<16, 256, 0, stream>>>(style, Wg, bg, Ws, bs, gp, sbuf);
    gn_stats_kernel<<<64, 256, 0, stream>>>(x, gnm, gnr);
    pos_kernel<<<HW, 256, 0, stream>>>(Wp1, bp1, Wp2, bp2, pos);
    tok_kv_kernel<<<BATCH * NTOK, 256, 0, stream>>>(toks, sbuf, ln_t_g, ln_t_b,
                                                    Wk, bk, Wv, bv, kb, vb);
    qin_kernel<<<BATCH * (HW / 32), 256, 0, stream>>>(x, pos, gnm, gnr,
                                                      ln_q_g, ln_q_b, qin);
    // q = q_in @ Wq + bq -> bf16
    gemm_bf16_kernel<<<dim3(MROWS / 64, 2), 256, 0, stream>>>(
        qin, WQT, bq, nullptr, nullptr, qbuf, 256, 256, /*flags*/2);
    // attention -> sc (bf16)
    attn_kernel<<<dim3(HW / 64, BATCH * NHEAD), 128, 0, stream>>>(qbuf, kb, vb, scb);
    // o = sc @ Wo + bo -> bf16
    gemm_bf16_kernel<<<dim3(MROWS / 64, 2), 256, 0, stream>>>(
        scb, WOT, bo, nullptr, nullptr, obuf, 256, 256, 2);
    // LN(o) -> bf16 (reuses q_in buffer)
    ln_rows_kernel<<<MROWS / 8, 256, 0, stream>>>(obuf, ln_f_g, ln_f_b, hln);
    // h = silu(LN(o) @ Wf1 + bf1) -> bf16
    gemm_bf16_kernel<<<dim3(MROWS / 64, 4), 256, 0, stream>>>(
        hln, WF1T, bf1, nullptr, nullptr, hmid, 512, 256, 1 | 2);
    // F = h @ Wf2 + bf2 + o  -> f32
    gemm_bf16_kernel<<<dim3(MROWS / 64, 2), 256, 0, stream>>>(
        hmid, WF2T, bf2, obuf, Fbuf, nullptr, 256, 512, 4);
    // final combine -> d_out (f32 NCHW)
    out_kernel<<<BATCH * (HW / 32), 256, 0, stream>>>(x, Fbuf, gnm, gnr, gp, gamma,
                                                      (float*)d_out);
}
